// MyBertPooler_55825984913418
// MI455X (gfx1250) — matrix operand order repaired and checked by reading, not compiled, not run on hardware
//
#include <hip/hip_runtime.h>
#include <hip/hip_bf16.h>
#include <math.h>

#define B_  16
#define S_  8192
#define H_  1024
#define NC  16            // S-chunks for the pooling pass
#define SCH (S_ / NC)     // 512 s-iterations per block

typedef __attribute__((ext_vector_type(16))) _Float16 v16h;
typedef __attribute__((ext_vector_type(8)))  _Float16 v8h;
typedef __attribute__((ext_vector_type(8)))  float    v8f;
typedef __attribute__((ext_vector_type(4)))  float    v4f;

// ---------------------------------------------------------------------------
// Kernel 1: single-pass online-softmax partial reduction over an S-chunk.
// Block = 256 threads, each thread owns 4 consecutive h via float4 loads, so
// one block covers all H=1024 channels of one (b, s-chunk). Grid = B*NC.
// Streams 512MB once with non-temporal loads (bigger than the 192MB L2).
// ---------------------------------------------------------------------------
__global__ __launch_bounds__(256) void pool_partial_kernel(
    const float* __restrict__ x,
    float* __restrict__ pm, float* __restrict__ pz, float* __restrict__ ps)
{
    const int b    = blockIdx.x / NC;
    const int chnk = blockIdx.x % NC;
    const int h4   = threadIdx.x;   // handles h = 4*h4 .. 4*h4+3

    const v4f* __restrict__ p =
        (const v4f*)(x + (size_t)b * S_ * H_ + (size_t)chnk * SCH * H_) + h4;

    float m0 = -INFINITY, m1 = -INFINITY, m2 = -INFINITY, m3 = -INFINITY;
    float z0 = 0.f, z1 = 0.f, z2 = 0.f, z3 = 0.f;
    float n0 = 0.f, n1 = 0.f, n2 = 0.f, n3 = 0.f;

#define ONLINE_UPD(v_, m, z, n)                                        \
    if ((v_) > (m)) {                                                  \
        float sc_ = __expf((m) - (v_));                                \
        (z) = (z) * sc_ + 1.0f;                                        \
        (n) = (n) * sc_ + (v_);                                        \
        (m) = (v_);                                                    \
    } else {                                                           \
        float e_ = __expf((v_) - (m));                                 \
        (z) += e_;                                                     \
        (n) += (v_) * e_;                                              \
    }

#pragma unroll 4
    for (int s = 0; s < SCH; ++s) {
        v4f v = __builtin_nontemporal_load(p + (size_t)s * (H_ / 4));
        ONLINE_UPD(v.x, m0, z0, n0)
        ONLINE_UPD(v.y, m1, z1, n1)
        ONLINE_UPD(v.z, m2, z2, n2)
        ONLINE_UPD(v.w, m3, z3, n3)
    }
#undef ONLINE_UPD

    const size_t o = ((size_t)(chnk * B_ + b) * H_ + 4 * (size_t)h4);
    *(v4f*)(pm + o) = (v4f){m0, m1, m2, m3};
    *(v4f*)(pz + o) = (v4f){z0, z1, z2, z3};
    *(v4f*)(ps + o) = (v4f){n0, n1, n2, n3};
}

// ---------------------------------------------------------------------------
// Kernel 2: combine NC partial (m,Z,num) triplets per (b,h), emit pooled as
// f16 A-matrix [16 x 1024] for the WMMA GEMM.
// ---------------------------------------------------------------------------
__global__ __launch_bounds__(256) void pool_combine_kernel(
    const float* __restrict__ pm, const float* __restrict__ pz,
    const float* __restrict__ ps, _Float16* __restrict__ Ah)
{
    const int idx    = blockIdx.x * 256 + threadIdx.x;   // over B*H = 16384
    const int stride = B_ * H_;

    float m = -INFINITY;
#pragma unroll
    for (int c = 0; c < NC; ++c)
        m = fmaxf(m, pm[(size_t)c * stride + idx]);

    float z = 0.f, n = 0.f;
#pragma unroll
    for (int c = 0; c < NC; ++c) {
        const size_t o = (size_t)c * stride + idx;
        float sc = __expf(pm[o] - m);
        z += pz[o] * sc;
        n += ps[o] * sc;
    }
    Ah[idx] = (_Float16)(n / z);
}

// ---------------------------------------------------------------------------
// Kernel 3: W f32 -> f16 (row-major [n][k], 1M elements, trivial).
// ---------------------------------------------------------------------------
__global__ __launch_bounds__(256) void wconvert_kernel(
    const float* __restrict__ W, _Float16* __restrict__ Wh)
{
    const size_t i = ((size_t)blockIdx.x * 256 + threadIdx.x) * 4;
    v4f v = *(const v4f*)(W + i);
    Wh[i + 0] = (_Float16)v.x;
    Wh[i + 1] = (_Float16)v.y;
    Wh[i + 2] = (_Float16)v.z;
    Wh[i + 3] = (_Float16)v.w;
}

// ---------------------------------------------------------------------------
// Kernel 4: out[16,1024] = tanh(pooled @ W^T + bias) via v_wmma_f32_16x16x32_f16.
// M=16 is exactly one WMMA tile; one wave per 16-wide N tile (64 waves total),
// K-loop of 32 over H=1024.
//
// A 16x32 as loaded here: M = lane&15; halves 0..7 -> K = lg*16+0..7,
//                          halves 8..15 -> K = lg*16+8..15   (lg = lane>>4)
// B 32x16 as loaded here (same order as A): N = lane&15; halves 0..15 -> K = lg*16 + 0..15
// C/D f32 layout: row r -> (M = r + 8*lg, N = lane&15)
// ---------------------------------------------------------------------------
__global__ __launch_bounds__(128) void gemm_tanh_kernel(
    const _Float16* __restrict__ Ah, const _Float16* __restrict__ Wh,
    const float* __restrict__ bias, float* __restrict__ out)
{
    const int wave = blockIdx.x * (blockDim.x >> 5) + (threadIdx.x >> 5); // ntile 0..63
    const int lane = threadIdx.x & 31;
    const int nl   = lane & 15;
    const int lg   = lane >> 4;
    const int n    = wave * 16 + nl;

    const _Float16* __restrict__ arow = Ah + (size_t)nl * H_ + lg * 16;
    const _Float16* __restrict__ brow = Wh + (size_t)n  * H_ + lg * 16;

    v8f c = {};
    for (int k0 = 0; k0 < H_; k0 += 32) {
        v8h alo = *(const v8h*)(arow + k0);        // K = lg*16 + 0..7
        v8h ahi = *(const v8h*)(arow + k0 + 8);    // K = lg*16 + 8..15
        v16h a;
#pragma unroll
        for (int i = 0; i < 8; ++i) { a[i] = alo[i]; a[i + 8] = ahi[i]; }

        v16h bt = *(const v16h*)(brow + k0);       // K = lg*16 + 0..15

        c = __builtin_amdgcn_wmma_f32_16x16x32_f16(
                /*neg_a=*/false, a, /*neg_b=*/false, bt,
                /*c_mod=*/(short)0, c, /*reuse_a=*/false, /*reuse_b=*/false);
    }

    const float bb = bias[n];
#pragma unroll
    for (int r = 0; r < 8; ++r) {
        const int mrow = r + 8 * lg;               // batch index 0..15
        out[(size_t)mrow * H_ + n] = tanhf(c[r] + bb);
    }
}

// ---------------------------------------------------------------------------
extern "C" void kernel_launch(void* const* d_in, const int* in_sizes, int n_in,
                              void* d_out, int out_size, void* d_ws, size_t ws_size,
                              hipStream_t stream) {
    const float* x    = (const float*)d_in[0];   // [B,S,H] f32
    const float* W    = (const float*)d_in[1];   // [H,H]   f32
    const float* bias = (const float*)d_in[2];   // [H]     f32
    float*       out  = (float*)d_out;           // [B,H]   f32

    // Workspace layout (≈5.1 MB): NC partial triplets + f16 A + f16 W
    const size_t PBH = (size_t)NC * B_ * H_;
    float*    pm = (float*)d_ws;
    float*    pz = pm + PBH;
    float*    ps = pz + PBH;
    _Float16* Ah = (_Float16*)(ps + PBH);
    _Float16* Wh = Ah + (size_t)B_ * H_;

    // Phase 1: stream 512MB once, online-softmax partials (bandwidth bound).
    pool_partial_kernel<<<B_ * NC, 256, 0, stream>>>(x, pm, pz, ps);
    // Phase 2: combine partials -> pooled[16,1024] in f16.
    pool_combine_kernel<<<(B_ * H_) / 256, 256, 0, stream>>>(pm, pz, ps, Ah);
    // Phase 3: W -> f16.
    wconvert_kernel<<<(H_ * H_) / (256 * 4), 256, 0, stream>>>(W, Wh);
    // Phase 4: WMMA GEMM (16x1024x1024) + bias + tanh.
    gemm_tanh_kernel<<<16, 128, 0, stream>>>(Ah, Wh, bias, out);
}
